// MultiHeadAttention_80204219285950
// MI455X (gfx1250) — matrix operand order repaired and checked by reading, not compiled, not run on hardware
//
#include <hip/hip_runtime.h>
#include <hip/hip_bf16.h>
#include <stdint.h>

// ---------------------------------------------------------------------------
// MHA forward for MI455X (gfx1250, wave32, WMMA).
// All GEMMs use v_wmma_f32_16x16x32_bf16 (bf16 in, fp32 accumulate).
// Register-staged pipeline; LDS layouts chosen so every WMMA fragment is
// built from wide contiguous LDS loads with zero sub-word repacking:
//   A: row-major, 80B rows  -> fragment = 2 x ds_load_b128
//   B: K-pair interleaved   -> fragment = 8 x b32 (merged ds_load_2addr_b32)
// ---------------------------------------------------------------------------

typedef __bf16 bf16_t;
typedef bf16_t v16bf __attribute__((ext_vector_type(16)));
typedef bf16_t v8bf  __attribute__((ext_vector_type(8)));
typedef bf16_t v4bf  __attribute__((ext_vector_type(4)));
typedef bf16_t v2bf  __attribute__((ext_vector_type(2)));
typedef float  v8f   __attribute__((ext_vector_type(8)));
typedef int    v8i   __attribute__((ext_vector_type(8)));
typedef int    v4i   __attribute__((ext_vector_type(4)));

#define B_     16
#define S_     1024
#define E_     768
#define H_     8
#define D_     96

static __device__ __forceinline__ bf16_t f2bf(float f) { return (bf16_t)f; }

// Block tile 128x128, K-step 32, 256 threads (8 waves), each wave -> 32x64
// (2x4 tiles of 16x16).
#define BMT   128
#define BNT   128
#define BKT   32
#define ASTR  40    // A LDS stride: 80B rows (16B aligned, 20*m mod 64 distinct)
#define BNP   132   // B LDS pair-row stride (in n units); pair rows 528B apart

// Bs flat index of element (k, n): ((k>>1)*BNP + n)*2 + (k&1)

// AMODE: 0 = A fp32 row-major [M,lda], 1 = A bf16 row-major
// BMODE: 0 = B fp32 row-major [K,ldb], 1 = B bf16 row-major [K,ldb],
//        2 = B bf16, source row-major [N,ldb] accessed transposed (B = srcT)
// CBF  : true -> store bf16, false -> store fp32
// BIAS : add bias[col]
// GUARD: bounds-check N (only needed when N % 128 != 0; requires N % 8 == 0)
template <int AMODE, int BMODE, bool CBF, bool BIAS, bool GUARD>
__global__ __launch_bounds__(256)
void gemm_wmma(const void* __restrict__ Ap, const void* __restrict__ Bp,
               const float* __restrict__ bias, void* __restrict__ Cp,
               int M, int N, int K, int lda, int ldb, int ldc,
               long long aOuter, long long aInner,
               long long bOuter, long long bInner,
               long long cOuter, long long cInner,
               int innerN, float scale)
{
    __shared__ __align__(16) bf16_t As[BMT * ASTR];          // 10.0 KB
    __shared__ __align__(16) bf16_t Bs[(BKT / 2) * BNP * 2]; //  8.25 KB

    const int tid      = threadIdx.x;
    const int lane     = tid & 31;
    const int wave     = tid >> 5;
    const int wm       = wave & 3;        // 0..3 -> 32-row group
    const int wn       = wave >> 2;       // 0..1 -> 64-col group
    const int laneHalf = lane >> 4;       // 0 or 1
    const int lane16   = lane & 15;

    const int blockRow = blockIdx.y * BMT;
    const int blockCol = blockIdx.x * BNT;

    const long long z  = (long long)blockIdx.z;
    const long long zo = z / innerN;
    const long long zi = z - zo * innerN;

    const float*  Af  = (const float*) Ap + zo * aOuter + zi * aInner;
    const bf16_t* Abf = (const bf16_t*)Ap + zo * aOuter + zi * aInner;
    const float*  Bf  = (const float*) Bp + zo * bOuter + zi * bInner;
    const bf16_t* Bbf = (const bf16_t*)Bp + zo * bOuter + zi * bInner;
    float*        Cf  = (float*)      Cp + zo * cOuter + zi * cInner;
    bf16_t*       Cb  = (bf16_t*)     Cp + zo * cOuter + zi * cInner;

    // ---- staging registers for the next K-tile ----
    float4 aF[4]; v8bf aH[2];
    float4 bF[4]; v8bf bH[2];

    auto loadA = [&](int k0) {
        if constexpr (AMODE == 0) {
            // 128x32 fp32: 4 coalesced float4 per thread
#pragma unroll
            for (int i = 0; i < 4; ++i) {
                int c  = tid + i * 256;           // 0..1023 float4 slots
                int r  = c >> 3;
                int k4 = (c & 7) << 2;
                aF[i] = *reinterpret_cast<const float4*>(
                    Af + (size_t)(blockRow + r) * lda + k0 + k4);
            }
        } else {
            // 128x32 bf16: 2 x (8 bf16) per thread
#pragma unroll
            for (int i = 0; i < 2; ++i) {
                int c  = tid + i * 256;           // 0..511 8-elem chunks
                int r  = c >> 2;
                int k8 = (c & 3) << 3;
                aH[i] = *reinterpret_cast<const v8bf*>(
                    Abf + (size_t)(blockRow + r) * lda + k0 + k8);
            }
        }
    };
    auto storeA = [&]() {
        if constexpr (AMODE == 0) {
#pragma unroll
            for (int i = 0; i < 4; ++i) {
                int c  = tid + i * 256;
                int r  = c >> 3;
                int k4 = (c & 7) << 2;
                v4bf p;
                p[0] = f2bf(aF[i].x); p[1] = f2bf(aF[i].y);
                p[2] = f2bf(aF[i].z); p[3] = f2bf(aF[i].w);
                *reinterpret_cast<v4bf*>(&As[r * ASTR + k4]) = p;
            }
        } else {
#pragma unroll
            for (int i = 0; i < 2; ++i) {
                int c  = tid + i * 256;
                int r  = c >> 2;
                int k8 = (c & 3) << 3;
                *reinterpret_cast<v8bf*>(&As[r * ASTR + k8]) = aH[i];  // b128
            }
        }
    };
    auto loadB = [&](int k0) {
        if constexpr (BMODE == 0) {
            // rows 2p,2p+1 x 8 cols per thread (p = tid>>4, n8 = (tid&15)*8)
            int p  = tid >> 4;
            int n8 = (tid & 15) << 3;
            const float* r0 = Bf + (size_t)(k0 + 2 * p)     * ldb + blockCol + n8;
            const float* r1 = Bf + (size_t)(k0 + 2 * p + 1) * ldb + blockCol + n8;
            bF[0] = *reinterpret_cast<const float4*>(r0);
            bF[1] = *reinterpret_cast<const float4*>(r0 + 4);
            bF[2] = *reinterpret_cast<const float4*>(r1);
            bF[3] = *reinterpret_cast<const float4*>(r1 + 4);
        } else if constexpr (BMODE == 1) {
            // rows 2p,2p+1 x 8 cols, chunk-granular N guard
            int p  = tid >> 4;
            int n8 = (tid & 15) << 3;
            int gC = blockCol + n8;
            if (GUARD && gC + 8 > N) {
#pragma unroll
                for (int j = 0; j < 8; ++j) { bH[0][j] = (bf16_t)0.0f; bH[1][j] = (bf16_t)0.0f; }
            } else {
                bH[0] = *reinterpret_cast<const v8bf*>(
                    Bbf + (size_t)(k0 + 2 * p) * ldb + gC);
                bH[1] = *reinterpret_cast<const v8bf*>(
                    Bbf + (size_t)(k0 + 2 * p + 1) * ldb + gC);
            }
        } else {
            // transposed source [N, ldb]: 8 consecutive k at one n per chunk
#pragma unroll
            for (int i = 0; i < 2; ++i) {
                int c  = tid + i * 256;
                int n  = c >> 2;
                int d8 = (c & 3) << 3;
                bH[i] = *reinterpret_cast<const v8bf*>(
                    Bbf + (size_t)(blockCol + n) * ldb + k0 + d8);
            }
        }
    };
    auto storeB = [&]() {
        if constexpr (BMODE == 0) {
            int p  = tid >> 4;
            int n8 = (tid & 15) << 3;
            const float* r0 = (const float*)&bF[0];   // k even, cols n8..n8+7
            const float* r1 = (const float*)&bF[2];   // k odd
            v8bf q0, q1;
#pragma unroll
            for (int j = 0; j < 4; ++j) {
                q0[2 * j]     = f2bf(r0[j]);     // cvt_pk pairs (k, k+1)
                q0[2 * j + 1] = f2bf(r1[j]);
                q1[2 * j]     = f2bf(r0[4 + j]);
                q1[2 * j + 1] = f2bf(r1[4 + j]);
            }
            *reinterpret_cast<v8bf*>(&Bs[(p * BNP + n8) * 2])     = q0; // b128
            *reinterpret_cast<v8bf*>(&Bs[(p * BNP + n8 + 4) * 2]) = q1; // b128
        } else if constexpr (BMODE == 1) {
            int p  = tid >> 4;
            int n8 = (tid & 15) << 3;
            v8bf q0, q1;
#pragma unroll
            for (int j = 0; j < 4; ++j) {
                q0[2 * j]     = bH[0][j];
                q0[2 * j + 1] = bH[1][j];
                q1[2 * j]     = bH[0][4 + j];
                q1[2 * j + 1] = bH[1][4 + j];
            }
            *reinterpret_cast<v8bf*>(&Bs[(p * BNP + n8) * 2])     = q0;
            *reinterpret_cast<v8bf*>(&Bs[(p * BNP + n8 + 4) * 2]) = q1;
        } else {
            // pairs along k are adjacent in the register -> 4 b32 stores
#pragma unroll
            for (int i = 0; i < 2; ++i) {
                int c  = tid + i * 256;
                int n  = c >> 2;
                int d8 = (c & 3) << 3;
#pragma unroll
                for (int j = 0; j < 4; ++j) {
                    v2bf pr;
                    pr[0] = bH[i][2 * j];
                    pr[1] = bH[i][2 * j + 1];
                    *reinterpret_cast<v2bf*>(
                        &Bs[(((d8 >> 1) + j) * BNP + n) * 2]) = pr;
                }
            }
        }
    };

    v8f acc[2][4];
#pragma unroll
    for (int tm = 0; tm < 2; ++tm)
#pragma unroll
        for (int tn = 0; tn < 4; ++tn)
#pragma unroll
            for (int r = 0; r < 8; ++r)
                acc[tm][tn][r] = 0.0f;

    const int nSteps = K / BKT;          // all K here are multiples of 32

    // prologue: tile 0 -> regs -> LDS
    loadA(0); loadB(0);
    storeA(); storeB();
    __syncthreads();

    for (int kt = 0; kt < nSteps; ++kt) {
        const bool hasNext = (kt + 1 < nSteps);
        if (hasNext) { loadA((kt + 1) * BKT); loadB((kt + 1) * BKT); }
        if (kt + 2 < nSteps) {
            // speculative L2 prefetch of the tile after next
            size_t nb = (size_t)(blockRow + (tid >> 1)) * lda + (kt + 2) * BKT;
            if constexpr (AMODE == 0) __builtin_prefetch((const void*)(Af + nb), 0, 0);
            else                      __builtin_prefetch((const void*)(Abf + nb), 0, 0);
        }

        // ---- A fragments: two contiguous 16B halves per tile ----
        v16bf afrag[2];
#pragma unroll
        for (int tm = 0; tm < 2; ++tm) {
            int m = wm * 32 + tm * 16 + lane16;
            const bf16_t* arow = &As[m * ASTR + (laneHalf << 4)];
            v4i lo = *reinterpret_cast<const v4i*>(arow);        // K kb..kb+7
            v4i hi = *reinterpret_cast<const v4i*>(arow + 8);    // K kb+8..+15
            v8i ai;
#pragma unroll
            for (int j = 0; j < 4; ++j) { ai[j] = lo[j]; ai[4 + j] = hi[j]; }
            afrag[tm] = __builtin_bit_cast(v16bf, ai);
        }
        // ---- B fragments: 8 b32 pair-words (ds_load_2addr_b32 pairs) ----
        v16bf bfrag[4];
#pragma unroll
        for (int tn = 0; tn < 4; ++tn) {
            int n = wn * 64 + tn * 16 + lane16;
            v8i bi;
#pragma unroll
            for (int v = 0; v < 8; ++v) {
                int p = (laneHalf << 3) + v;
                bi[v] = *reinterpret_cast<const int*>(&Bs[(p * BNP + n) * 2]);
            }
            bfrag[tn] = __builtin_bit_cast(v16bf, bi);
        }
#pragma unroll
        for (int tm = 0; tm < 2; ++tm)
#pragma unroll
            for (int tn = 0; tn < 4; ++tn)
                acc[tm][tn] = __builtin_amdgcn_wmma_f32_16x16x32_bf16(
                    false, afrag[tm], false, bfrag[tn],
                    (short)0, acc[tm][tn], false, false);

        if (hasNext) {
            __syncthreads();            // all waves done reading current tile
            storeA(); storeB();
            __syncthreads();            // new tile visible
        }
    }

    // ---- epilogue: C layout M = r + 8*(lane>=16), N = lane%16 ----
#pragma unroll
    for (int tm = 0; tm < 2; ++tm)
#pragma unroll
        for (int tn = 0; tn < 4; ++tn) {
            int col = blockCol + wn * 64 + tn * 16 + lane16;
            int rB  = blockRow + wm * 32 + tm * 16 + laneHalf * 8;
            if (GUARD && col >= N) continue;
            float b = BIAS ? bias[col] : 0.0f;
#pragma unroll
            for (int r = 0; r < 8; ++r) {
                int gR = rB + r;
                float v = acc[tm][tn][r] * scale + b;
                if constexpr (CBF) Cb[(size_t)gR * ldc + col] = f2bf(v);
                else               Cf[(size_t)gR * ldc + col] = v;
            }
        }
}

// Softmax over the HEAD axis (faithful to reference: softmax(weight, axis=1)).
// attn layout (b, h, q, k); for each (b,q,k) normalize across the 8 heads.
__global__ __launch_bounds__(256)
void softmax_heads(float* __restrict__ attn, long long plane,
                   long long batchStride, long long total)
{
    long long t = (long long)blockIdx.x * 256 + threadIdx.x;
    if (t >= total) return;
    long long b   = t / plane;
    long long rem = t - b * plane;
    float* p = attn + b * batchStride + rem;

    float v[H_];
    float mx = -3.4e38f;
#pragma unroll
    for (int h = 0; h < H_; ++h) { v[h] = p[(long long)h * plane]; mx = fmaxf(mx, v[h]); }
    float s = 0.0f;
#pragma unroll
    for (int h = 0; h < H_; ++h) { v[h] = __expf(v[h] - mx); s += v[h]; }
    float inv = 1.0f / s;
#pragma unroll
    for (int h = 0; h < H_; ++h) p[(long long)h * plane] = v[h] * inv;
}

extern "C" void kernel_launch(void* const* d_in, const int* in_sizes, int n_in,
                              void* d_out, int out_size, void* d_ws, size_t ws_size,
                              hipStream_t stream)
{
    const float* x  = (const float*)d_in[0];
    const float* Wq = (const float*)d_in[1];
    const float* bq = (const float*)d_in[2];
    const float* Wk = (const float*)d_in[3];
    const float* bk = (const float*)d_in[4];
    const float* Wv = (const float*)d_in[5];
    const float* bv = (const float*)d_in[6];
    const float* Wo = (const float*)d_in[7];
    const float* bo = (const float*)d_in[8];

    const int M = B_ * S_;                        // 16384 rows
    float* out  = (float*)d_out;                  // [16,1024,768]
    float* attn = (float*)d_out + (size_t)M * E_; // [16,8,1024,1024]

    bf16_t* Qbf = (bf16_t*)d_ws;
    bf16_t* Kbf = Qbf + (size_t)M * E_;
    bf16_t* Vbf = Kbf + (size_t)M * E_;
    bf16_t* Cbf = Vbf + (size_t)M * E_;           // context, bf16

    dim3 blk(256, 1, 1);

    // 1) Q/K/V projections: x(fp32) @ W(fp32) + b -> bf16
    dim3 gP(E_ / BNT, M / BMT, 1);
    gemm_wmma<0, 0, true, true, false><<<gP, blk, 0, stream>>>(
        x, Wq, bq, Qbf, M, E_, E_, E_, E_, E_, 0, 0, 0, 0, 0, 0, 1, 1.0f);
    gemm_wmma<0, 0, true, true, false><<<gP, blk, 0, stream>>>(
        x, Wk, bk, Kbf, M, E_, E_, E_, E_, E_, 0, 0, 0, 0, 0, 0, 1, 1.0f);
    gemm_wmma<0, 0, true, true, false><<<gP, blk, 0, stream>>>(
        x, Wv, bv, Vbf, M, E_, E_, E_, E_, E_, 0, 0, 0, 0, 0, 0, 1, 1.0f);

    // 2) scores: per (b,h)  Q @ K^T * scale -> fp32 into attention output
    const float scaling = 0.10206207261596577f;   // 96^-0.5
    dim3 gS(S_ / BNT, S_ / BMT, B_ * H_);
    gemm_wmma<1, 2, false, false, false><<<gS, blk, 0, stream>>>(
        Qbf, Kbf, nullptr, attn,
        S_, S_, D_, E_, E_, S_,
        (long long)S_ * E_, (long long)D_,            // A: +b*S*E, +h*D
        (long long)S_ * E_, (long long)D_,            // B: +b*S*E, +h*D
        (long long)H_ * S_ * S_, (long long)S_ * S_,  // C: +b*H*S*S, +h*S*S
        H_, scaling);

    // 3) softmax across heads, in place on d_out attention region
    long long total = (long long)B_ * S_ * S_;    // 16.7M (b,q,k) triples
    softmax_heads<<<(unsigned)(total / 256), blk, 0, stream>>>(
        attn, (long long)S_ * S_, (long long)H_ * S_ * S_, total);

    // 4) context: per (b,h)  attn(fp32) @ V(bf16) -> bf16 [b,q,h*96+d]
    dim3 gC(1, S_ / BMT, B_ * H_);
    gemm_wmma<0, 1, true, false, true><<<gC, blk, 0, stream>>>(
        attn, Vbf, nullptr, Cbf,
        S_, D_, S_, S_, E_, E_,
        (long long)H_ * S_ * S_, (long long)S_ * S_,  // A: +b*H*S*S, +h*S*S
        (long long)S_ * E_, (long long)D_,            // B: +b*S*E, +h*D
        (long long)S_ * E_, (long long)D_,            // C: +b*S*E, +h*D
        H_, 1.0f);

    // 5) out = context @ Wo + bo -> fp32
    gemm_wmma<1, 0, false, true, false><<<gP, blk, 0, stream>>>(
        Cbf, Wo, bo, out, M, E_, E_, E_, E_, E_, 0, 0, 0, 0, 0, 0, 1, 1.0f);
}
